// _NonLocalBlockND_33483565039876
// MI455X (gfx1250) — matrix operand order repaired and checked by reading, not compiled, not run on hardware
//
#include <hip/hip_runtime.h>
#include <hip/hip_bf16.h>

// ---------------------------------------------------------------------------
// NonLocalBlockND (3-branch attention) for MI455X / gfx1250.
// All GEMMs run on v_wmma_f32_16x16x32_f16 (wave32 WMMA, f32 accumulate).
// Global->LDS staging uses CDNA5 GLOBAL_LOAD_ASYNC_TO_LDS_B128 with
// double-buffered LDS tiles and s_wait_asynccnt pipelining.
// Workspace requirement: ~78.4 MB.
// ---------------------------------------------------------------------------

typedef __attribute__((ext_vector_type(16))) _Float16 v16h;
typedef __attribute__((ext_vector_type(8)))  _Float16 v8h;
typedef __attribute__((ext_vector_type(8)))  float    v8f;

#define LDSS 48   // LDS row stride in halves (96B: multiple of 32B)

// Generic tiled WMMA GEMM:  acc[M x N] = A[M x K] (row-major, f16)
//                                      x Bt[N x K] (row-major, f16)^T
// Block tile 128x128, 8 waves (2 M x 4 N), each wave 64x32 = 4x2 WMMA tiles.
// epi: 0 = f16 store (+optional row/col bias)          [projections]
//      1 = f32 store                                   [attention scores S]
//      2 = f32 accumulate (+=)                         [y += P*g]
//      3 = f16 store of (acc + colBias + res[c][r]) * dwW + dwB  [cat buffer]
//      4 = f32 store of res[r][c] + p*(acc + rowBias)  [final pointwise]
__global__ __launch_bounds__(256) void gemm_wmma(
    const _Float16* __restrict__ A,  int lda, long aBatch,
    const _Float16* __restrict__ Bt, int ldb, long bBatch,
    void* __restrict__ Cout, int ldc, long cBatch,
    int K, int epi,
    const float* __restrict__ rowBias,
    const float* __restrict__ colBias,
    const float* __restrict__ res, int resLd, long resBatch,
    const float* __restrict__ dwW, const float* __restrict__ dwB,
    const float* __restrict__ pScale)
{
    // double-buffered tiles: [A buf0][A buf1][B buf0][B buf1]
    __shared__ alignas(128) _Float16 smem[4 * 128 * LDSS];
    constexpr unsigned TILEH  = 128 * LDSS;        // halves per buffer
    constexpr unsigned BUFB   = TILEH * 2;         // bytes per buffer (12288)
    const unsigned ldsBase = (unsigned)(unsigned long long)(void*)&smem[0];

    const int bz = blockIdx.z;
    A  += (long)bz * aBatch;
    Bt += (long)bz * bBatch;
    if (res) res += (long)bz * resBatch;

    const int tid   = threadIdx.x;
    const int wid   = tid >> 5;
    const int lane  = tid & 31;
    const int waveM = wid >> 2;     // 0..1
    const int waveN = wid & 3;      // 0..3
    const int half  = lane >> 4;    // 0/1
    const int l16   = lane & 15;

    const int blockM = blockIdx.y * 128;
    const int blockN = blockIdx.x * 128;

    // staging map: thread -> (row 0..127, 16-half chunk 0/1) = 32B/thread/tile
    const int srow = tid >> 1;
    const int soff = (tid & 1) * 16;                 // halves
    const unsigned ldsStage = (unsigned)(srow * LDSS + soff) * 2;  // bytes
    const long aOff0 = ((long)(blockM + srow) * lda + soff) * 2;   // bytes
    const long bOff0 = ((long)(blockN + srow) * ldb + soff) * 2;

    // Issue async global->LDS copies for K-tile t (4 x b128 per thread).
    auto issueTile = [&](int t) {
        const unsigned bsel = (unsigned)(t & 1) * BUFB;
        const unsigned la = ldsBase + bsel + ldsStage;
        const unsigned lb = ldsBase + 2 * BUFB + bsel + ldsStage;
        const unsigned ga = (unsigned)(aOff0 + (long)t * 64);
        const unsigned gb = (unsigned)(bOff0 + (long)t * 64);
        asm volatile(
            "global_load_async_to_lds_b128 %0, %2, %4\n\t"
            "global_load_async_to_lds_b128 %1, %3, %4"
            :: "v"(la), "v"(la + 16), "v"(ga), "v"(ga + 16), "s"(A)
            : "memory");
        asm volatile(
            "global_load_async_to_lds_b128 %0, %2, %4\n\t"
            "global_load_async_to_lds_b128 %1, %3, %4"
            :: "v"(lb), "v"(lb + 16), "v"(gb), "v"(gb + 16), "s"(Bt)
            : "memory");
    };

    v8f acc[4][2];
#pragma unroll
    for (int i = 0; i < 4; ++i)
#pragma unroll
        for (int j = 0; j < 2; ++j) acc[i][j] = (v8f)0.f;

    const int T = K >> 5;           // K-tiles of 32
    issueTile(0);

    for (int t = 0; t < T; ++t) {
        __syncthreads();            // readers of buf[(t+1)&1] (iter t-1) done
        if (t + 1 < T) {
            issueTile(t + 1);
            // tile t's 4 async ops complete in order once <=4 outstanding
            asm volatile("s_wait_asynccnt 0x4" ::: "memory");
        } else {
            asm volatile("s_wait_asynccnt 0x0" ::: "memory");
        }
        __syncthreads();            // tile t staged by every wave

        const _Float16* As = smem + (unsigned)(t & 1) * TILEH;
        const _Float16* Bs = smem + 2 * TILEH + (unsigned)(t & 1) * TILEH;

        // A fragment (16x32 f16): lanes 0-15 -> M=l16, K 0-7 & 16-23;
        //                          lanes16-31 -> M=l16, K 8-15 & 24-31.
        v16h afr[4];
#pragma unroll
        for (int sm = 0; sm < 4; ++sm) {
            const _Float16* pa = &As[(waveM * 64 + sm * 16 + l16) * LDSS];
            v8h lo = *(const v8h*)(pa + half * 8);
            v8h hi = *(const v8h*)(pa + 16 + half * 8);
            afr[sm] = __builtin_shufflevector(lo, hi,
                0,1,2,3,4,5,6,7,8,9,10,11,12,13,14,15);
        }
        // B fragment (32x16): lane -> column N=l16; lanes 0-15 hold K 0-7 & 16-23,
        //                     lanes 16-31 hold K 8-15 & 24-31 (same as A)
        v16h bfr[2];
#pragma unroll
        for (int sn = 0; sn < 2; ++sn) {
            const _Float16* pb = &Bs[(waveN * 32 + sn * 16 + l16) * LDSS];
            v8h lo = *(const v8h*)(pb + half * 8);
            v8h hi = *(const v8h*)(pb + 16 + half * 8);
            bfr[sn] = __builtin_shufflevector(lo, hi,
                0,1,2,3,4,5,6,7,8,9,10,11,12,13,14,15);
        }
#pragma unroll
        for (int sm = 0; sm < 4; ++sm)
#pragma unroll
            for (int sn = 0; sn < 2; ++sn)
                acc[sm][sn] = __builtin_amdgcn_wmma_f32_16x16x32_f16(
                    false, afr[sm], false, bfr[sn],
                    (short)0, acc[sm][sn], false, false);
    }

    const float p = (epi == 4 && pScale) ? pScale[0] : 0.f;

    // C/D layout: VGPR r, lanes 0-15 -> M=r, N=lane; lanes 16-31 -> M=r+8.
#pragma unroll
    for (int sm = 0; sm < 4; ++sm) {
#pragma unroll
        for (int sn = 0; sn < 2; ++sn) {
#pragma unroll
            for (int r = 0; r < 8; ++r) {
                const int gr = blockM + waveM * 64 + sm * 16 + r + 8 * half;
                const int gc = blockN + waveN * 32 + sn * 16 + l16;
                float v = acc[sm][sn][r];
                if (epi == 0) {
                    if (rowBias) v += rowBias[gr];
                    if (colBias) v += colBias[gc];
                    ((_Float16*)Cout)[(long)bz * cBatch + (long)gr * ldc + gc] =
                        (_Float16)v;
                } else if (epi == 1) {
                    ((float*)Cout)[(long)bz * cBatch + (long)gr * ldc + gc] = v;
                } else if (epi == 2) {
                    float* o = (float*)Cout + (long)bz * cBatch +
                               (long)gr * ldc + gc;
                    *o = *o + v;
                } else if (epi == 3) {
                    v += colBias[gc] + res[(long)gc * resLd + gr];
                    v = v * dwW[gc] + dwB[gc];
                    ((_Float16*)Cout)[(long)bz * cBatch + (long)gr * ldc + gc] =
                        (_Float16)v;
                } else { // epi == 4
                    float r0 = res[(long)gr * resLd + gc];
                    ((float*)Cout)[(long)bz * cBatch + (long)gr * ldc + gc] =
                        r0 + p * (v + rowBias[gr]);
                }
            }
        }
    }
}

// Row softmax: one block per row of S (n=2304 = 9*256), writes f16 probs.
__global__ __launch_bounds__(256) void softmax_rows(
    const float* __restrict__ S, _Float16* __restrict__ P, int ncols)
{
    __shared__ float red[256];
    const int row = blockIdx.x;
    const float* s = S + (long)row * ncols;
    float vals[9];
    float mx = -1e30f;
#pragma unroll
    for (int i = 0; i < 9; ++i) {
        vals[i] = s[threadIdx.x + i * 256];
        mx = fmaxf(mx, vals[i]);
    }
    red[threadIdx.x] = mx; __syncthreads();
    for (int st = 128; st > 0; st >>= 1) {
        if (threadIdx.x < st)
            red[threadIdx.x] = fmaxf(red[threadIdx.x], red[threadIdx.x + st]);
        __syncthreads();
    }
    mx = red[0]; __syncthreads();
    float sum = 0.f;
#pragma unroll
    for (int i = 0; i < 9; ++i) { vals[i] = __expf(vals[i] - mx); sum += vals[i]; }
    red[threadIdx.x] = sum; __syncthreads();
    for (int st = 128; st > 0; st >>= 1) {
        if (threadIdx.x < st) red[threadIdx.x] += red[threadIdx.x + st];
        __syncthreads();
    }
    const float inv = 1.f / red[0];
    _Float16* pr = P + (long)row * ncols;
#pragma unroll
    for (int i = 0; i < 9; ++i)
        pr[threadIdx.x + i * 256] = (_Float16)(vals[i] * inv);
}

// (C,n) f32 -> (n,C) f16 transpose-convert, 32x32 LDS tiles, grid.z = batch.
__global__ __launch_bounds__(256) void transpose_convert(
    const float* __restrict__ src, _Float16* __restrict__ dst, int Cdim, int n)
{
    __shared__ float T[32][33];
    const int b  = blockIdx.z;
    const int i0 = blockIdx.x * 32, c0 = blockIdx.y * 32;
    const int tx = threadIdx.x & 31, ty = threadIdx.x >> 5;
    src += (long)b * Cdim * n;
    dst += (long)b * n * Cdim;
#pragma unroll
    for (int r = 0; r < 4; ++r)
        T[ty + r * 8][tx] = src[(long)(c0 + ty + r * 8) * n + i0 + tx];
    __syncthreads();
#pragma unroll
    for (int r = 0; r < 4; ++r)
        dst[(long)(i0 + ty + r * 8) * Cdim + c0 + tx] =
            (_Float16)T[tx][ty + r * 8];
}

__global__ void f32_to_f16(const float* __restrict__ s,
                           _Float16* __restrict__ d, long count)
{
    long i = (long)blockIdx.x * blockDim.x + threadIdx.x;
    if (i < count) d[i] = (_Float16)s[i];
}

// ---------------------------------------------------------------------------
extern "C" void kernel_launch(void* const* d_in, const int* in_sizes, int n_in,
                              void* d_out, int out_size, void* d_ws, size_t ws_size,
                              hipStream_t stream)
{
    (void)in_sizes; (void)n_in; (void)out_size; (void)ws_size;
    const int Bn = 4, Cc = 256, N = 2304 /*48*48*/, CI = 128, Q = 768, O = 1024;

    const float* x[3]  = { (const float*)d_in[0], (const float*)d_in[1],
                           (const float*)d_in[2] };
    const float* x_in  = (const float*)d_in[3];
    const float* wproj[7] = { (const float*)d_in[4],  (const float*)d_in[6],
                              (const float*)d_in[8],  (const float*)d_in[10],
                              (const float*)d_in[12], (const float*)d_in[14],
                              (const float*)d_in[16] };       // g,th,ph,th1,ph1,th2,ph2
    const float* g_b   = (const float*)d_in[5];
    const float* th_b[3] = { (const float*)d_in[7],  (const float*)d_in[11],
                             (const float*)d_in[15] };
    const float* ph_b[3] = { (const float*)d_in[9],  (const float*)d_in[13],
                             (const float*)d_in[17] };
    const float* Ww[3] = { (const float*)d_in[18], (const float*)d_in[20],
                           (const float*)d_in[22] };
    const float* Wb[3] = { (const float*)d_in[19], (const float*)d_in[21],
                           (const float*)d_in[23] };
    const float* dw_w  = (const float*)d_in[24];
    const float* dw_b  = (const float*)d_in[25];
    const float* pw_w  = (const float*)d_in[26];
    const float* pw_b  = (const float*)d_in[27];
    const float* psc   = (const float*)d_in[28];
    float* out = (float*)d_out;

    // workspace layout (byte offsets, all 512-aligned)
    char* ws = (char*)d_ws;
    _Float16* xh    = (_Float16*)(ws);                         // 3*B*N*C f16
    _Float16* whp   = (_Float16*)(ws + 14155776);              // 7 * 128*256 f16
    _Float16* whW   = (_Float16*)(ws + 14614528);              // 3 * 256*128 f16
    _Float16* pwh   = (_Float16*)(ws + 14811136);              // 1024*768 f16
    _Float16* theta = (_Float16*)(ws + 16384000);              // [j][b][N][CI] f16
    _Float16* phiT  = (_Float16*)(ws + 23461888);              // [j][b][N][CI] f16
    _Float16* gT    = (_Float16*)(ws + 30539776);              // [k][b][CI][N] f16
    float*    S     = (float*)   (ws + 37617664);              // N*N f32 (reused)
    _Float16* P     = (_Float16*)(ws + 58851328);              // N*N f16 (reused)
    float*    y     = (float*)   (ws + 69468160);              // [k][N][CI] f32 (per batch)
    _Float16* yh    = (_Float16*)(ws + 73007104);              // [k][N][CI] f16
    _Float16* catd  = (_Float16*)(ws + 74776576);              // [N][768] f16

    // 1) transpose-convert inputs: (C,n) f32 -> (n,C) f16
    for (int t = 0; t < 3; ++t)
        transpose_convert<<<dim3(N / 32, Cc / 32, Bn), 256, 0, stream>>>(
            x[t], xh + (size_t)t * Bn * N * Cc, Cc, N);

    // 2) weight conversions to f16
    for (int k = 0; k < 7; ++k)
        f32_to_f16<<<128, 256, 0, stream>>>(wproj[k], whp + (size_t)k * 32768, 32768);
    for (int k = 0; k < 3; ++k)
        f32_to_f16<<<128, 256, 0, stream>>>(Ww[k], whW + (size_t)k * 32768, 32768);
    f32_to_f16<<<3072, 256, 0, stream>>>(pw_w, pwh, (long)O * Q);

    // 3) projections (batched over grid.z)
    for (int j = 0; j < 3; ++j) {
        const _Float16* xh_j = xh + (size_t)j * Bn * N * Cc;
        // theta_j: [N x CI] = xh_j * th_w^T
        gemm_wmma<<<dim3(1, 18, Bn), 256, 0, stream>>>(
            xh_j, Cc, (long)N * Cc,
            whp + (size_t)(1 + 2 * j) * 32768, Cc, 0,
            theta + (size_t)j * Bn * N * CI, CI, (long)N * CI,
            Cc, 0, nullptr, th_b[j], nullptr, 0, 0, nullptr, nullptr, nullptr);
        // phi_j^T: [N x CI] = xh_j * ph_w^T
        gemm_wmma<<<dim3(1, 18, Bn), 256, 0, stream>>>(
            xh_j, Cc, (long)N * Cc,
            whp + (size_t)(2 + 2 * j) * 32768, Cc, 0,
            phiT + (size_t)j * Bn * N * CI, CI, (long)N * CI,
            Cc, 0, nullptr, ph_b[j], nullptr, 0, 0, nullptr, nullptr, nullptr);
        // g_j^T: [CI x N] = g_w * xh_j^T
        gemm_wmma<<<dim3(18, 1, Bn), 256, 0, stream>>>(
            whp, Cc, 0,
            xh_j, Cc, (long)N * Cc,
            gT + (size_t)j * Bn * CI * N, N, (long)CI * N,
            Cc, 0, g_b, nullptr, nullptr, 0, 0, nullptr, nullptr, nullptr);
    }

    // 4) per-batch attention + mixing + fused tail
    for (int b = 0; b < Bn; ++b) {
        hipMemsetAsync(y, 0, (size_t)3 * N * CI * sizeof(float), stream);
        for (int j = 0; j < 3; ++j) {
            const _Float16* th = theta + ((size_t)j * Bn + b) * N * CI;
            const _Float16* ph = phiT  + ((size_t)j * Bn + b) * N * CI;
            // S = theta_j * phi_j   (N x N, f32)
            gemm_wmma<<<dim3(18, 18, 1), 256, 0, stream>>>(
                th, CI, 0, ph, CI, 0, S, N, 0,
                CI, 1, nullptr, nullptr, nullptr, 0, 0, nullptr, nullptr, nullptr);
            softmax_rows<<<N, 256, 0, stream>>>(S, P, N);
            // y_k += P * g_k   (N x CI, f32 accumulate)
            for (int k = 0; k < 3; ++k)
                gemm_wmma<<<dim3(1, 18, 1), 256, 0, stream>>>(
                    P, N, 0,
                    gT + ((size_t)k * Bn + b) * CI * N, N, 0,
                    y + (size_t)k * N * CI, CI, 0,
                    N, 2, nullptr, nullptr, nullptr, 0, 0, nullptr, nullptr, nullptr);
        }
        f32_to_f16<<<3456, 256, 0, stream>>>(y, yh, (long)3 * N * CI);
        // cat (transposed, dw applied): catd[i][k*256+c] =
        //     dw_w*( y_k[i]*W_k^T + W_b + x_k[c][i] ) + dw_b
        for (int k = 0; k < 3; ++k)
            gemm_wmma<<<dim3(2, 18, 1), 256, 0, stream>>>(
                yh + (size_t)k * N * CI, CI, 0,
                whW + (size_t)k * 32768, CI, 0,
                catd + (size_t)k * 256, Q, 0,
                CI, 3, nullptr, Wb[k],
                x[k] + (size_t)b * Cc * N, N, 0,
                dw_w + (size_t)k * 256, dw_b + (size_t)k * 256, nullptr);
        // out = x_in + p*(pw_w * catd^T + pw_b)
        gemm_wmma<<<dim3(18, 8, 1), 256, 0, stream>>>(
            pwh, Q, 0, catd, Q, 0,
            out + (size_t)b * O * N, N, 0,
            Q, 4, pw_b, nullptr,
            x_in + (size_t)b * O * N, N, 0, nullptr, nullptr, psc);
    }
}
